// DiscreteDiffusion_30004641530329
// MI455X (gfx1250) — matrix operand order repaired and checked by reading, not compiled, not run on hardware
//
#include <hip/hip_runtime.h>

// ---------------------------------------------------------------------------
// DiscreteDiffusion CRPS loss, MI455X (gfx1250, wave32, WMMA)
//   B=8, N=32768 (T16*H32*W64), D=32, E=8  -> scalar f32 output
// Phase A: per-batch exact top-k threshold via 128KB-LDS radix select
// Phase B: fused masked-GEMM (v_wmma_f32_16x16x32_f16, transposed so each
//          lane owns a full 8-member ensemble) + kernel-CRPS + masked mean
// Round 2 changes: per-batch 1/rate_corr hoisted to an LDS table (removes
//   v_div_* + fmodf from the hot loop); wave-uniform skip of all-visible
//   tiles (their loss weight is identically zero).
// ---------------------------------------------------------------------------

typedef __attribute__((ext_vector_type(16))) _Float16 h16_t;
typedef __attribute__((ext_vector_type(8)))  _Float16 h8_t;
typedef __attribute__((ext_vector_type(8)))  float    f8_t;

#define N_TOK 32768
#define DIM   32
#define BATCH 8

__global__ void kzero(float* out) {
  if (threadIdx.x == 0 && blockIdx.x == 0) out[0] = 0.0f;
}

// ---------------- Phase A: per-batch top-k selection ------------------------
// One workgroup (1024 thr = 32 waves) per batch. All 32768 order-preserving
// u32 keys live in dynamic LDS (128 KB; CDNA5 allows 320 KB/WGP).
__global__ __launch_bounds__(1024) void topk_select(
    const float* __restrict__ u_g,  const float* __restrict__ dir_t,
    const float* __restrict__ dir_h, const float* __restrict__ dir_w,
    const float* __restrict__ u_rate, unsigned char* __restrict__ vis) {
  extern __shared__ unsigned int keys[];          // [32768]
  __shared__ unsigned int hist[256];
  __shared__ unsigned int s_prefix, s_need, s_ticket;

  const int b   = blockIdx.x;
  const int tid = threadIdx.x;

  // ws = gumbel(u_g) + dir_t[t] + dir_h[h] + dir_w[w], mapped to sortable u32
  for (int i = tid; i < N_TOK; i += 1024) {
    float u  = u_g[(size_t)b * N_TOK + i];
    float g  = -logf(-logf(u));
    int   w  = i & 63, h = (i >> 6) & 31, t = i >> 11;
    float ws = g + dir_t[b * 16 + t] + dir_h[b * 32 + h] + dir_w[b * 64 + w];
    unsigned bits = __float_as_uint(ws);
    keys[i] = (bits & 0x80000000u) ? ~bits : (bits | 0x80000000u);
  }

  // k_b from stratified rates: ((u_rate + b/7) % 1) * N, clipped to [1, N-1]
  float rate = fmodf(u_rate[0] + (float)b * (1.0f / 7.0f), 1.0f);
  int   kb   = (int)(32768.0f * rate);
  kb = min(max(kb, 1), N_TOK - 1);

  if (tid == 0) { s_prefix = 0u; s_need = (unsigned)kb; s_ticket = 0u; }
  __syncthreads();

  // 4-round MSB-first radix select: exact k-th largest key
  for (int round = 0; round < 4; ++round) {
    const int      shift  = 24 - 8 * round;
    const unsigned maskHi = (round == 0) ? 0u : (0xFFFFFFFFu << (shift + 8));
    if (tid < 256) hist[tid] = 0u;
    __syncthreads();
    const unsigned prefix = s_prefix;
    for (int i = tid; i < N_TOK; i += 1024) {
      unsigned key = keys[i];
      if ((key & maskHi) == prefix)
        atomicAdd(&hist[(key >> shift) & 255u], 1u);
    }
    __syncthreads();
    if (tid == 0) {
      unsigned need = s_need, cum = 0u;
      for (int bin = 255; bin >= 0; --bin) {
        unsigned c2 = cum + hist[bin];
        if (c2 >= need) {
          s_need   = need - cum;
          s_prefix = prefix | ((unsigned)bin << shift);
          break;
        }
        cum = c2;
      }
    }
    __syncthreads();
  }

  const unsigned T = s_prefix, needEq = s_need;
  // visible = key > T, plus `needEq` of the ties (LDS ticket)
  for (int i = tid; i < N_TOK; i += 1024) {
    unsigned key = keys[i];
    unsigned char v;
    if      (key > T)  v = 1;
    else if (key == T) v = (atomicAdd(&s_ticket, 1u) < needEq) ? 1 : 0;
    else               v = 0;
    vis[(size_t)b * N_TOK + i] = v;
  }
}

// ---------------- Phase B: fused WMMA GEMM + CRPS + masked reduction --------
#define CSWAP(a, b) { float lo_ = fminf(a, b), hi_ = fmaxf(a, b); a = lo_; b = hi_; }

__global__ __launch_bounds__(256) void gemm_crps(
    const float* __restrict__ tokens, const float* __restrict__ Wm,
    const float* __restrict__ bnet,   const float* __restrict__ u_rate,
    const unsigned char* __restrict__ vis, float* __restrict__ out) {
  // W^T in f16 (A-operand source), bias, per-batch weight table, and
  // per-wave f32 token tiles
  __shared__ __align__(16) _Float16 Wt[256 * 32];   // 16 KB: Wt[ch][k] = W[k][ch]
  __shared__ __align__(16) float    bsh[256];       // 1 KB
  __shared__ __align__(16) float    inv_tab[8];     // N/(N-k_b) per batch
  __shared__ __align__(16) float    tile[8 * 512];  // 16 KB: per-wave 16tok x 32dim

  const int tid  = threadIdx.x;
  const int wave = tid >> 5;
  const int lane = tid & 31;

  for (int idx = tid; idx < 256 * 32; idx += 256) {
    int c = idx >> 5, k = idx & 31;
    Wt[idx] = (_Float16)Wm[k * 256 + c];
  }
  if (tid < 256) bsh[tid] = bnet[tid];
  if (tid < 8) {  // hoisted per-batch rate correction (was a per-tile v_div)
    float rate = fmodf(u_rate[0] + (float)tid * (1.0f / 7.0f), 1.0f);
    int   kb   = min(max((int)(32768.0f * rate), 1), N_TOK - 1);
    inv_tab[tid] = 32768.0f / (float)(N_TOK - kb);
  }
  __syncthreads();

  const int      gwave   = blockIdx.x * 8 + wave;
  const int      nwaves  = gridDim.x * 8;
  const int      tok_l   = lane & 15;        // token within 16-wide tile
  const int      kh      = lane >> 4;        // K-half (0: K0-15, 1: K16-31)
  const unsigned ldsBase = (unsigned)(unsigned long long)(&tile[wave * 512]);

  float partial = 0.0f;
  const int TILES = BATCH * (N_TOK / 16);    // 16384 tiles, wave-uniform loop

  for (int t = gwave; t < TILES; t += nwaves) {
    const int b    = t >> 11;
    const int tok0 = (t & 2047) << 4;

    // ---- async-stage 16 tokens x 32 dims (2 KB, contiguous) into LDS ----
    const unsigned long long gbase =
        (unsigned long long)(tokens + (((size_t)b << 15) + tok0) * DIM);
    asm volatile("s_wait_dscnt 0" ::: "memory");   // prior tile's LDS reads done
#pragma unroll
    for (int r = 0; r < 4; ++r) {
      unsigned           off = (unsigned)(r * 512 + lane * 16);
      unsigned           la  = ldsBase + off;
      unsigned long long ga  = gbase + off;
      asm volatile("global_load_async_to_lds_b128 %0, %1, off"
                   :: "v"(la), "v"(ga) : "memory");
    }

    unsigned char v = vis[((size_t)b << 15) + tok0 + tok_l];

    // all 16 tokens visible -> every lane's loss weight is 0: skip tile.
    // (__any is wave-uniform; EXEC stays all-ones inside, as WMMA requires)
    if (!__any(v == 0)) continue;

    float vscale  = v ? 1.0f : 0.0f;            // masks GEMM input row
    float wgt     = (1.0f - vscale) * inv_tab[b]; // loss weight (~visible)

    asm volatile("s_wait_asynccnt 0" ::: "memory");

    // ---- B operand: x^T tile (32xK x 16tok), f16, visibility-masked ----
    const float4* t4 = (const float4*)(&tile[wave * 512 + tok_l * 32 + kh * 16]);
    h16_t bfrag;
#pragma unroll
    for (int q = 0; q < 4; ++q) {
      float4 f = t4[q];
      bfrag[4 * q + 0] = (_Float16)(f.x * vscale);
      bfrag[4 * q + 1] = (_Float16)(f.y * vscale);
      bfrag[4 * q + 2] = (_Float16)(f.z * vscale);
      bfrag[4 * q + 3] = (_Float16)(f.w * vscale);
    }

    // ---- 16 channel-tiles: WMMA then lane-local CRPS over the 8-ensemble --
#pragma unroll
    for (int c = 0; c < 16; ++c) {
      const _Float16* arow = &Wt[(c * 16 + tok_l) * 32 + kh * 16];
      h8_t lo = *(const h8_t*)arow;
      h8_t hi = *(const h8_t*)(arow + 8);
      h16_t afrag;
#pragma unroll
      for (int j = 0; j < 8; ++j) { afrag[j] = lo[j]; afrag[j + 8] = hi[j]; }

      f8_t acc = {};
      acc = __builtin_amdgcn_wmma_f32_16x16x32_f16(
          false, afrag, false, bfrag, (short)0, acc, false, false);

      const int d  = 2 * c + kh;                  // this lane's d index
      float tv = tile[wave * 512 + tok_l * 32 + d];
      const float* bn = &bsh[16 * c + 8 * kh];    // bias for e = 0..7

      float e0 = acc[0] + bn[0], e1 = acc[1] + bn[1];
      float e2 = acc[2] + bn[2], e3 = acc[3] + bn[3];
      float e4 = acc[4] + bn[4], e5 = acc[5] + bn[5];
      float e6 = acc[6] + bn[6], e7 = acc[7] + bn[7];

      float t1 = (fabsf(tv - e0) + fabsf(tv - e1) + fabsf(tv - e2) +
                  fabsf(tv - e3) + fabsf(tv - e4) + fabsf(tv - e5) +
                  fabsf(tv - e6) + fabsf(tv - e7)) * 0.125f;

      // Batcher odd-even sort-of-8 (19 comparators), lane-local
      CSWAP(e0, e1) CSWAP(e2, e3) CSWAP(e4, e5) CSWAP(e6, e7)
      CSWAP(e0, e2) CSWAP(e1, e3) CSWAP(e4, e6) CSWAP(e5, e7)
      CSWAP(e1, e2) CSWAP(e5, e6)
      CSWAP(e0, e4) CSWAP(e1, e5) CSWAP(e2, e6) CSWAP(e3, e7)
      CSWAP(e2, e4) CSWAP(e3, e5)
      CSWAP(e1, e2) CSWAP(e3, e4) CSWAP(e5, e6)

      float t2 = (-7.0f * e0 - 5.0f * e1 - 3.0f * e2 - e3 +
                  e4 + 3.0f * e5 + 5.0f * e6 + 7.0f * e7) * (1.0f / 32.0f);

      partial += wgt * (t1 - 0.5f * t2);
    }
  }

  // drain any async loads issued for a skipped final tile
  asm volatile("s_wait_asynccnt 0" ::: "memory");

  // wave32 reduction + single f32 atomic per wave
#pragma unroll
  for (int o = 16; o > 0; o >>= 1) partial += __shfl_down(partial, o, 32);
  if (lane == 0)
    atomicAdd(out, partial * (1.0f / (8.0f * 32768.0f * 32.0f)));
}

// ---------------------------------------------------------------------------
extern "C" void kernel_launch(void* const* d_in, const int* in_sizes, int n_in,
                              void* d_out, int out_size, void* d_ws, size_t ws_size,
                              hipStream_t stream) {
  const float* tokens = (const float*)d_in[0];
  const float* Wm     = (const float*)d_in[1];
  const float* bnet   = (const float*)d_in[2];
  const float* u_g    = (const float*)d_in[3];
  const float* dir_t  = (const float*)d_in[4];
  const float* dir_h  = (const float*)d_in[5];
  const float* dir_w  = (const float*)d_in[6];
  const float* u_rate = (const float*)d_in[7];
  float*       out    = (float*)d_out;
  unsigned char* vis  = (unsigned char*)d_ws;   // 256 KB of workspace

  kzero<<<1, 32, 0, stream>>>(out);
  topk_select<<<BATCH, 1024, N_TOK * sizeof(unsigned int), stream>>>(
      u_g, dir_t, dir_h, dir_w, u_rate, vis);
  gemm_crps<<<512, 256, 0, stream>>>(tokens, Wm, bnet, u_rate, vis, out);
}
